// EdgeTransformer_GNN_29781303231141
// MI455X (gfx1250) — matrix operand order repaired and checked by reading, not compiled, not run on hardware
//
#include <hip/hip_runtime.h>
#include <hip/hip_bf16.h>
#include <math.h>

// Problem constants (match reference)
#define NN 50000      // nodes
#define NE 400000     // edges
#define D  128        // hidden dim (= H*C = D_IN)
#define DE 32         // edge feature dim
#define NH 8          // heads
#define CH 16         // channels per head
#define NL 4          // layers

typedef __attribute__((ext_vector_type(16))) __bf16 v16bf;
typedef __attribute__((ext_vector_type(8)))  __bf16 v8bf;
typedef __attribute__((ext_vector_type(4)))  __bf16 v4bf;
typedef __attribute__((ext_vector_type(8)))  float  v8f;

// WMMA 16x16x32 bf16 fragment layout used here (wave32; contiguous k, same for A and B, not AMD's table order):
//  A (16x32 MxK): lane -> M=lane&15; half=lane>>4.
//    elements 0..15 = K [half*16 .. half*16+15]           (one contiguous 16-run)
//  B (32x16 KxN): lane -> N=lane&15; half selects K run:
//    elements 0..15 = K [half*16 .. half*16+15]           (one contiguous 16-run)
//  C/D (16x16 f32): VGPR r -> M = r + 8*half, N = lane&15.

__device__ __forceinline__ v16bf loadA(const __bf16* __restrict__ base) {
  // base points at K-offset half*16 of this lane's row (+k0); one contiguous 32B run
  const v8bf a0 = *(const v8bf*)(base);
  const v8bf a1 = *(const v8bf*)(base + 8);
  return __builtin_shufflevector(a0, a1, 0, 1, 2, 3, 4, 5, 6, 7,
                                 8, 9, 10, 11, 12, 13, 14, 15);
}
__device__ __forceinline__ v16bf loadB(const __bf16* __restrict__ base) {
  // one contiguous 32B run
  const v8bf b0 = *(const v8bf*)(base);
  const v8bf b1 = *(const v8bf*)(base + 8);
  return __builtin_shufflevector(b0, b1, 0, 1, 2, 3, 4, 5, 6, 7,
                                 8, 9, 10, 11, 12, 13, 14, 15);
}

__device__ __forceinline__ void atomicMaxFloat(float* addr, float val) {
  // valid given init = -inf (ordered-int trick)
  if (val >= 0.0f) atomicMax((int*)addr, __float_as_int(val));
  else             atomicMin((unsigned int*)addr, (unsigned int)__float_as_int(val));
}

// ---- fp32 -> bf16 staging (4 elements/thread) ------------------------------
__global__ void conv_bf16_kernel(const float* __restrict__ src, __bf16* __restrict__ dst, int n4)
{
  const int t = blockIdx.x * blockDim.x + threadIdx.x;
  if (t >= n4) return;
  const float4 v = ((const float4*)src)[t];
  v4bf o;
  o[0] = (__bf16)v.x; o[1] = (__bf16)v.y; o[2] = (__bf16)v.z; o[3] = (__bf16)v.w;
  *(v4bf*)(dst + (size_t)t * 4) = o;
}

// ---- Fused node projection: Q = hWq^T+bq, K, V, Skip (one A frag, 4 WMMAs/kstep)
__global__ void __launch_bounds__(256) node_proj_kernel(
    const __bf16* __restrict__ h16,
    const __bf16* __restrict__ wq, const float* __restrict__ bq,
    const __bf16* __restrict__ wk, const float* __restrict__ bk,
    const __bf16* __restrict__ wv, const float* __restrict__ bv,
    const __bf16* __restrict__ wsk, const float* __restrict__ bs,
    float* __restrict__ Q, float* __restrict__ Ko,
    float* __restrict__ V, float* __restrict__ S)
{
  const int lane = threadIdx.x & 31;
  const int wid  = blockIdx.x * (blockDim.x >> 5) + (threadIdx.x >> 5);
  const int TN = D / 16;                       // 8 column tiles
  if (wid >= (NN / 16) * TN) return;           // wave-uniform exit
  const int tm = wid / TN, tn = wid % TN;
  const int m0 = tm * 16, n0 = tn * 16;
  const int r = lane & 15, half = lane >> 4;

  v8f aq, ak, av, as2;
  const float vq = bq[n0 + r], vk = bk[n0 + r], vv = bv[n0 + r], vs = bs[n0 + r];
#pragma unroll
  for (int i = 0; i < 8; ++i) { aq[i] = vq; ak[i] = vk; av[i] = vv; as2[i] = vs; }

  const __bf16* aRow = h16 + (size_t)(m0 + r) * D + half * 16;  // + k0
  const __bf16* bRowQ = wq  + (size_t)(n0 + r) * D + half * 16; // + k0
  const __bf16* bRowK = wk  + (size_t)(n0 + r) * D + half * 16;
  const __bf16* bRowV = wv  + (size_t)(n0 + r) * D + half * 16;
  const __bf16* bRowS = wsk + (size_t)(n0 + r) * D + half * 16;

#pragma unroll
  for (int ks = 0; ks < D / 32; ++ks) {
    const int k0 = ks * 32;
    const v16bf A = loadA(aRow + k0);
    v16bf B;
    B  = loadB(bRowQ + k0);
    aq = __builtin_amdgcn_wmma_f32_16x16x32_bf16(false, A, false, B, (short)0, aq, false, false);
    B  = loadB(bRowK + k0);
    ak = __builtin_amdgcn_wmma_f32_16x16x32_bf16(false, A, false, B, (short)0, ak, false, false);
    B  = loadB(bRowV + k0);
    av = __builtin_amdgcn_wmma_f32_16x16x32_bf16(false, A, false, B, (short)0, av, false, false);
    B  = loadB(bRowS + k0);
    as2 = __builtin_amdgcn_wmma_f32_16x16x32_bf16(false, A, false, B, (short)0, as2, false, false);
  }

  const int col = n0 + r;
#pragma unroll
  for (int rr = 0; rr < 8; ++rr) {
    const int row = m0 + rr + (half << 3);
    Q [(size_t)row * D + col] = aq[rr];
    Ko[(size_t)row * D + col] = ak[rr];
    V [(size_t)row * D + col] = av[rr];
    S [(size_t)row * D + col] = as2[rr];
  }
}

// ---- Edge projection: Ep = edge_feats @ We^T  (K=32 -> single WMMA per tile)
__global__ void __launch_bounds__(256) edge_proj_kernel(
    const __bf16* __restrict__ ef16, const __bf16* __restrict__ we16, float* __restrict__ Ep)
{
  const int lane = threadIdx.x & 31;
  const int wid  = blockIdx.x * (blockDim.x >> 5) + (threadIdx.x >> 5);
  if (wid >= NE / 16) return;                  // wave-uniform exit
  const int m0 = wid * 16;
  const int r = lane & 15, half = lane >> 4;

  const v16bf A = loadA(ef16 + (size_t)(m0 + r) * DE + half * 16);

#pragma unroll
  for (int tn = 0; tn < D / 16; ++tn) {
    const v16bf B = loadB(we16 + (size_t)(tn * 16 + r) * DE + half * 16);
    v8f acc = {0.f, 0.f, 0.f, 0.f, 0.f, 0.f, 0.f, 0.f};
    acc = __builtin_amdgcn_wmma_f32_16x16x32_bf16(false, A, false, B, (short)0, acc, false, false);
    const int col = tn * 16 + r;
#pragma unroll
    for (int rr = 0; rr < 8; ++rr)
      Ep[(size_t)(m0 + rr + (half << 3)) * D + col] = acc[rr];
  }
}

// ---- Per-layer init: agg=0, m=-inf, s=0
__global__ void init_kernel(float* __restrict__ agg, float* __restrict__ mb, float* __restrict__ sb)
{
  const size_t t = (size_t)blockIdx.x * blockDim.x + threadIdx.x;
  if (t < (size_t)NN * D) agg[t] = 0.0f;
  if (t < (size_t)NN * NH) { mb[t] = -INFINITY; sb[t] = 0.0f; }
}

// ---- alpha_raw = <q[dst], k[src]+e> / sqrt(C); running per-(dst,head) max
__global__ void alpha_raw_kernel(const float* __restrict__ Q, const float* __restrict__ K,
                                 const float* __restrict__ Ep,
                                 const int* __restrict__ src, const int* __restrict__ dst,
                                 float* __restrict__ alpha, float* __restrict__ mb)
{
  const int t = blockIdx.x * blockDim.x + threadIdx.x;
  if (t >= NE * NH) return;
  const int e = t >> 3, hh = t & 7;
  const int s = src[e], d = dst[e];
  const float4* qp = (const float4*)(Q  + (size_t)d * D + hh * CH);
  const float4* kp = (const float4*)(K  + (size_t)s * D + hh * CH);
  const float4* ep = (const float4*)(Ep + (size_t)e * D + hh * CH);
  float acc = 0.0f;
#pragma unroll
  for (int i = 0; i < 4; ++i) {
    const float4 q = qp[i], k = kp[i], ev = ep[i];
    acc += q.x * (k.x + ev.x) + q.y * (k.y + ev.y) + q.z * (k.z + ev.z) + q.w * (k.w + ev.w);
  }
  acc *= 0.25f;                                // 1/sqrt(16)
  alpha[t] = acc;
  atomicMaxFloat(&mb[(size_t)d * NH + hh], acc);
}

// ---- a = exp(raw - m); running per-(dst,head) sum
__global__ void alpha_exp_kernel(const int* __restrict__ dst, const float* __restrict__ mb,
                                 float* __restrict__ alpha, float* __restrict__ sb)
{
  const int t = blockIdx.x * blockDim.x + threadIdx.x;
  if (t >= NE * NH) return;
  const int e = t >> 3, hh = t & 7;
  const int d = dst[e];
  const float a = expf(alpha[t] - mb[(size_t)d * NH + hh]);
  alpha[t] = a;
  atomicAdd(&sb[(size_t)d * NH + hh], a);
}

// ---- normalize alpha; scatter (v[src]+e)*alpha into agg[dst]
__global__ void scatter_kernel(const int* __restrict__ src, const int* __restrict__ dst,
                               const float* __restrict__ V, const float* __restrict__ Ep,
                               const float* __restrict__ sb,
                               float* __restrict__ alpha, float* __restrict__ agg)
{
  const int t = blockIdx.x * blockDim.x + threadIdx.x;
  if (t >= NE * NH) return;
  const int e = t >> 3, hh = t & 7;
  const int s = src[e], d = dst[e];
  const float w = alpha[t] / (sb[(size_t)d * NH + hh] + 1e-16f);
  alpha[t] = w;
  const float4* vp = (const float4*)(V  + (size_t)s * D + hh * CH);
  const float4* ep = (const float4*)(Ep + (size_t)e * D + hh * CH);
  float* ap = agg + (size_t)d * D + hh * CH;
#pragma unroll
  for (int i = 0; i < 4; ++i) {
    const float4 v = vp[i], ev = ep[i];
    atomicAdd(ap + 4 * i + 0, (v.x + ev.x) * w);
    atomicAdd(ap + 4 * i + 1, (v.y + ev.y) * w);
    atomicAdd(ap + 4 * i + 2, (v.z + ev.z) * w);
    atomicAdd(ap + 4 * i + 3, (v.w + ev.w) * w);
  }
}

// ---- out = agg + skip ; exact-erf GELU for layers 0..2
__global__ void out_kernel(const float* __restrict__ agg, const float* __restrict__ S,
                           float* __restrict__ out, int applyGelu)
{
  const size_t t = (size_t)blockIdx.x * blockDim.x + threadIdx.x;
  if (t >= (size_t)NN * D) return;
  float x = agg[t] + S[t];
  if (applyGelu) x = 0.5f * x * (1.0f + erff(x * 0.7071067811865476f));
  out[t] = x;
}

extern "C" void kernel_launch(void* const* d_in, const int* in_sizes, int n_in,
                              void* d_out, int out_size, void* d_ws, size_t ws_size,
                              hipStream_t stream) {
  (void)in_sizes; (void)n_in; (void)out_size; (void)ws_size;
  const float* x   = (const float*)d_in[0];
  const float* ef  = (const float*)d_in[1];
  const float* Wq  = (const float*)d_in[2];
  const float* bq  = (const float*)d_in[3];
  const float* Wk  = (const float*)d_in[4];
  const float* bk  = (const float*)d_in[5];
  const float* Wv  = (const float*)d_in[6];
  const float* bv  = (const float*)d_in[7];
  const float* We  = (const float*)d_in[8];
  const float* Ws  = (const float*)d_in[9];
  const float* bs  = (const float*)d_in[10];
  const int*   ei  = (const int*)d_in[11];
  const int* src = ei;          // edge_index[0]
  const int* dst = ei + NE;     // edge_index[1]

  float* out = (float*)d_out;

  // workspace carve-up
  float* ws  = (float*)d_ws;
  float* Qb  = ws;                       // NN*D fp32 each
  float* Kb  = Qb  + (size_t)NN * D;
  float* Vb  = Kb  + (size_t)NN * D;
  float* Sb  = Vb  + (size_t)NN * D;
  float* Hb  = Sb  + (size_t)NN * D;     // layer activations
  float* Agg = Hb  + (size_t)NN * D;
  float* Mb  = Agg + (size_t)NN * D;     // NN*NH
  float* Sm  = Mb  + (size_t)NN * NH;
  float* Ep  = Sm  + (size_t)NN * NH;    // NE*D fp32
  __bf16* h16  = (__bf16*)(Ep + (size_t)NE * D); // NN*D bf16
  __bf16* ef16 = h16  + (size_t)NN * D;          // NE*DE bf16
  __bf16* wq16 = ef16 + (size_t)NE * DE;         // D*D bf16 each
  __bf16* wk16 = wq16 + (size_t)D * D;
  __bf16* wv16 = wk16 + (size_t)D * D;
  __bf16* ws16 = wv16 + (size_t)D * D;
  __bf16* we16 = ws16 + (size_t)D * D;           // D*DE bf16

  const int BLK = 256;
  const int gInit    = ((NN * D) + BLK - 1) / BLK;          // 25000
  const int gNodeWmm = ((NN / 16) * (D / 16) + 7) / 8;      // 3125 blocks (8 waves)
  const int gEdgeWmm = ((NE / 16) + 7) / 8;                 // 3125 blocks
  const int gEdge    = ((NE * NH) + BLK - 1) / BLK;         // 12500
  const int gOut     = gInit;
  const int gConvH   = ((NN * D / 4) + BLK - 1) / BLK;      // 6250
  const int gConvEF  = ((NE * DE / 4) + BLK - 1) / BLK;     // 12500
  const int gConvW   = ((D * D / 4) + BLK - 1) / BLK;       // 16
  const int gConvWe  = ((D * DE / 4) + BLK - 1) / BLK;      // 4

  // edge features are layer-invariant: stage to bf16 once per call
  hipLaunchKernelGGL(conv_bf16_kernel, dim3(gConvEF), dim3(BLK), 0, stream, ef, ef16, NE * DE / 4);

  for (int l = 0; l < NL; ++l) {
    const float* hin  = (l == 0) ? x : Hb;
    float*       hout = (l == NL - 1) ? out : Hb;
    float* alphaOut = out + (size_t)NN * D + (size_t)l * NE * NH;
    const size_t wOff = (size_t)l * D * D;
    const size_t bOff = (size_t)l * D;
    const size_t eOff = (size_t)l * D * DE;

    // stage this layer's activations + weights as bf16
    hipLaunchKernelGGL(conv_bf16_kernel, dim3(gConvH),  dim3(BLK), 0, stream, hin,       h16,  NN * D / 4);
    hipLaunchKernelGGL(conv_bf16_kernel, dim3(gConvW),  dim3(BLK), 0, stream, Wq + wOff, wq16, D * D / 4);
    hipLaunchKernelGGL(conv_bf16_kernel, dim3(gConvW),  dim3(BLK), 0, stream, Wk + wOff, wk16, D * D / 4);
    hipLaunchKernelGGL(conv_bf16_kernel, dim3(gConvW),  dim3(BLK), 0, stream, Wv + wOff, wv16, D * D / 4);
    hipLaunchKernelGGL(conv_bf16_kernel, dim3(gConvW),  dim3(BLK), 0, stream, Ws + wOff, ws16, D * D / 4);
    hipLaunchKernelGGL(conv_bf16_kernel, dim3(gConvWe), dim3(BLK), 0, stream, We + eOff, we16, D * DE / 4);

    hipLaunchKernelGGL(init_kernel, dim3(gInit), dim3(BLK), 0, stream, Agg, Mb, Sm);
    hipLaunchKernelGGL(node_proj_kernel, dim3(gNodeWmm), dim3(BLK), 0, stream,
                       h16, wq16, bq + bOff, wk16, bk + bOff,
                       wv16, bv + bOff, ws16, bs + bOff,
                       Qb, Kb, Vb, Sb);
    hipLaunchKernelGGL(edge_proj_kernel, dim3(gEdgeWmm), dim3(BLK), 0, stream,
                       ef16, we16, Ep);
    hipLaunchKernelGGL(alpha_raw_kernel, dim3(gEdge), dim3(BLK), 0, stream,
                       Qb, Kb, Ep, src, dst, alphaOut, Mb);
    hipLaunchKernelGGL(alpha_exp_kernel, dim3(gEdge), dim3(BLK), 0, stream,
                       dst, Mb, alphaOut, Sm);
    hipLaunchKernelGGL(scatter_kernel, dim3(gEdge), dim3(BLK), 0, stream,
                       src, dst, Vb, Ep, Sm, alphaOut, Agg);
    hipLaunchKernelGGL(out_kernel, dim3(gOut), dim3(BLK), 0, stream,
                       Agg, Sb, hout, (l < NL - 1) ? 1 : 0);
  }
}
